// EGNNContext_69458211111180
// MI455X (gfx1250) — compile-verified, matrix operand order checked by reading, not run on hardware
//
#include <hip/hip_runtime.h>
#include <hip/hip_bf16.h>
#include <math.h>

typedef _Float16 half_t;
typedef __attribute__((ext_vector_type(16))) _Float16 v16h;
typedef __attribute__((ext_vector_type(8)))  float    v8f;

#define NN_   384
#define EE_   3072
#define CIN_  4096
#define EMB_  200
#define DD_   512
#define NOBJ_ 151
#define NREL_ 51
#define K1_   4424
#define K1P_  4448
#define K2_   4808
#define K2P_  4832

static inline int cdiv_h(int a, int b) { return (a + b - 1) / b; }

// ---------------------------------------------------------------------------
// WMMA GEMM:  C[M,Nout] = act( A[M,K] @ B[Npad,K]^T + bias ), f16 in / f32 out
// Contract: M % 128 == 0, Npad % 128 == 0 (B zero-padded rows), K % 32 == 0,
// rows 16B-aligned. Store clipped to Nout. Block 256 thr = 8 waves (2x4),
// tile 128x128, wave tile 64x32 (acc 4x2 WMMA frags). Double-buffered LDS
// staged with global_load_async_to_lds_b128 (ASYNCcnt), 4 async/wave/batch.
// ---------------------------------------------------------------------------
#define TM 128
#define TN 128
#define KS 32
#define LPAD 40   // halfs per LDS row (80 B stride, 16B-aligned)

__global__ __launch_bounds__(256)
void gemm_f16(const half_t* __restrict__ A, const half_t* __restrict__ B,
              const float* __restrict__ bias, float* __restrict__ C,
              int M, int Nout, int K, int ldc, int act)
{
    __shared__ half_t As[2][TM * LPAD];
    __shared__ half_t Bs[2][TN * LPAD];

    const int tid  = threadIdx.x;
    const int lane = tid & 31;
    const int wave = tid >> 5;
    const int wm   = wave >> 2;      // 0..1  (64 rows each)
    const int wn   = wave & 3;       // 0..3  (32 cols each)
    const int bM   = blockIdx.y * TM;
    const int bN   = blockIdx.x * TN;
    const int hi   = lane >> 4;
    const int l15  = lane & 15;

    // per-thread staging geometry: 512 16B-chunks per 128x32 tile, 2/thread
    const int r0 = tid >> 2;                 // chunk row, i=0
    const int r1 = (tid + 256) >> 2;         // chunk row, i=1
    const int kc = (tid & 3) * 8;            // half offset within row

    v8f acc[4][2];
    {
        v8f z = {0.f, 0.f, 0.f, 0.f, 0.f, 0.f, 0.f, 0.f};
        #pragma unroll
        for (int mi = 0; mi < 4; ++mi)
            #pragma unroll
            for (int ni = 0; ni < 2; ++ni) acc[mi][ni] = z;
    }

    // async-stage one k-tile into buffer `buf` (exactly 4 async instrs / wave)
    auto stage = [&](int buf, int k0) {
        {
            unsigned la = (unsigned)(uintptr_t)&As[buf][r0 * LPAD + kc];
            const half_t* gp = A + (size_t)(bM + r0) * K + k0 + kc;
            asm volatile("global_load_async_to_lds_b128 %0, %1, off" :: "v"(la), "v"(gp) : "memory");
        }
        {
            unsigned la = (unsigned)(uintptr_t)&As[buf][r1 * LPAD + kc];
            const half_t* gp = A + (size_t)(bM + r1) * K + k0 + kc;
            asm volatile("global_load_async_to_lds_b128 %0, %1, off" :: "v"(la), "v"(gp) : "memory");
        }
        {
            unsigned la = (unsigned)(uintptr_t)&Bs[buf][r0 * LPAD + kc];
            const half_t* gp = B + (size_t)(bN + r0) * K + k0 + kc;
            asm volatile("global_load_async_to_lds_b128 %0, %1, off" :: "v"(la), "v"(gp) : "memory");
        }
        {
            unsigned la = (unsigned)(uintptr_t)&Bs[buf][r1 * LPAD + kc];
            const half_t* gp = B + (size_t)(bN + r1) * K + k0 + kc;
            asm volatile("global_load_async_to_lds_b128 %0, %1, off" :: "v"(la), "v"(gp) : "memory");
        }
    };

    stage(0, 0);
    const int nk = K / KS;
    for (int t = 0; t < nk; ++t) {
        const int cur = t & 1;
        if (t + 1 < nk) {
            stage(cur ^ 1, (t + 1) * KS);
            // previous batch (4 async) is done once <=4 remain (in-order completion)
            asm volatile("s_wait_asynccnt 0x4" ::: "memory");
        } else {
            asm volatile("s_wait_asynccnt 0x0" ::: "memory");
        }
        __syncthreads();

        // fragments: 16-bit A-matrix 16x32 lane layout
        v16h af[4], bf[2];
        #pragma unroll
        for (int mi = 0; mi < 4; ++mi) {
            const half_t* ap = &As[cur][(wm * 64 + mi * 16 + l15) * LPAD];
            #pragma unroll
            for (int v = 0; v < 8; ++v) {
                int kb = (v < 4) ? (2 * v + 8 * hi) : (16 + 2 * (v - 4) + 8 * hi);
                af[mi][2 * v]     = ap[kb];
                af[mi][2 * v + 1] = ap[kb + 1];
            }
        }
        #pragma unroll
        for (int ni = 0; ni < 2; ++ni) {
            const half_t* bp = &Bs[cur][(wn * 32 + ni * 16 + l15) * LPAD];
            #pragma unroll
            for (int v = 0; v < 8; ++v) {
                int kb = (v < 4) ? (2 * v + 8 * hi) : (16 + 2 * (v - 4) + 8 * hi);
                bf[ni][2 * v]     = bp[kb];
                bf[ni][2 * v + 1] = bp[kb + 1];
            }
        }
        #pragma unroll
        for (int mi = 0; mi < 4; ++mi)
            #pragma unroll
            for (int ni = 0; ni < 2; ++ni)
                acc[mi][ni] = __builtin_amdgcn_wmma_f32_16x16x32_f16(
                    false, af[mi], false, bf[ni], (short)0, acc[mi][ni], false, false);
        __syncthreads();
    }

    // store: f32 C/D layout (lane = col, 8 VGPR rows)
    #pragma unroll
    for (int mi = 0; mi < 4; ++mi) {
        #pragma unroll
        for (int ni = 0; ni < 2; ++ni) {
            int n = bN + wn * 32 + ni * 16 + l15;
            if (n >= Nout) continue;
            float bv = bias ? bias[n] : 0.0f;
            #pragma unroll
            for (int r = 0; r < 8; ++r) {
                int m = bM + wm * 64 + mi * 16 + r + 8 * hi;
                if (m < M) {
                    float v = acc[mi][ni][r] + bv;
                    if (act) v = fmaxf(v, 0.0f);
                    C[(size_t)m * ldc + n] = v;
                }
            }
        }
    }
}

// ---------------------------------------------------------------------------
// Elementwise / glue kernels
// ---------------------------------------------------------------------------
// f32 -> f16 with row (Mr->Mp) and column (K->Kp) zero padding
__global__ void k_f32_to_f16(const float* __restrict__ s, half_t* __restrict__ d,
                             int Mr, int Mp, int K, int Kp)
{
    size_t i = (size_t)blockIdx.x * blockDim.x + threadIdx.x;
    size_t tot = (size_t)Mp * Kp;
    if (i >= tot) return;
    int col = (int)(i % Kp);
    size_t row = i / Kp;
    float v = (row < (size_t)Mr && col < K) ? s[row * K + col] : 0.0f;
    d[i] = (half_t)v;
}

__global__ void k_zero_f32(float* __restrict__ p, size_t n)
{
    size_t i = (size_t)blockIdx.x * blockDim.x + threadIdx.x;
    if (i < n) p[i] = 0.0f;
}

__global__ void k_fill_i32(int* __restrict__ p, int v, size_t n)
{
    size_t i = (size_t)blockIdx.x * blockDim.x + threadIdx.x;
    if (i < n) p[i] = v;
}

// ph = pos_info @ pos_w1^T + b1   (384 x 32, K=9)
__global__ void k_pos1(const float* __restrict__ pi, const float* __restrict__ w,
                       const float* __restrict__ b, float* __restrict__ ph)
{
    int i = blockIdx.x * blockDim.x + threadIdx.x;
    if (i >= NN_ * 32) return;
    int r = i / 32, c = i % 32;
    float s = b[c];
    #pragma unroll
    for (int j = 0; j < 9; ++j) s += pi[r * 9 + j] * w[c * 9 + j];
    ph[i] = s;
}

// per-column mean / rsqrt(var+eps) over 384 rows (batch-dim layernorm)
__global__ void k_colstats(const float* __restrict__ ph, float* __restrict__ mu,
                           float* __restrict__ rsig)
{
    int c = threadIdx.x;
    if (c >= 32) return;
    float s = 0.f, s2 = 0.f;
    for (int r = 0; r < NN_; ++r) { float v = ph[r * 32 + c]; s += v; s2 += v * v; }
    float m = s / NN_;
    float var = s2 / NN_ - m * m;
    mu[c] = m;
    rsig[c] = rsqrtf(var + 1e-5f);
}

__global__ void k_posnorm(const float* __restrict__ ph, const float* __restrict__ mu,
                          const float* __restrict__ rsig, const float* __restrict__ g,
                          const float* __restrict__ be, float* __restrict__ o)
{
    int i = blockIdx.x * blockDim.x + threadIdx.x;
    if (i >= NN_ * 32) return;
    int c = i % 32;
    o[i] = (ph[i] - mu[c]) * rsig[c] * g[c] + be[c];
}

// pose = relu(phn @ pos_w2^T + b2)  (384 x 128, K=32)
__global__ void k_pos2(const float* __restrict__ phn, const float* __restrict__ w,
                       const float* __restrict__ b, float* __restrict__ o)
{
    int i = blockIdx.x * blockDim.x + threadIdx.x;
    if (i >= NN_ * 128) return;
    int r = i / 128, c = i % 128;
    float s = b[c];
    #pragma unroll 8
    for (int j = 0; j < 32; ++j) s += phn[r * 32 + j] * w[c * 32 + j];
    o[i] = fmaxf(s, 0.0f);
}

// Xcat = [x | obj_embed1[lab] | pos_embed | 0pad]  (384 x 4448) f16
__global__ void k_build_xcat(const float* __restrict__ x, const int* __restrict__ lab,
                             const float* __restrict__ emb, const float* __restrict__ pose,
                             half_t* __restrict__ d)
{
    size_t i = (size_t)blockIdx.x * blockDim.x + threadIdx.x;
    if (i >= (size_t)NN_ * K1P_) return;
    int c = (int)(i % K1P_);
    int r = (int)(i / K1P_);
    float v = 0.f;
    if (c < CIN_)               v = x[(size_t)r * CIN_ + c];
    else if (c < CIN_ + EMB_)   v = emb[lab[r] * EMB_ + (c - CIN_)];
    else if (c < K1_)           v = pose[r * 128 + (c - CIN_ - EMB_)];
    d[i] = (half_t)v;
}

// Xcat2 = [x | node_states | obj_embed2[lab] | 0pad]  (384 x 4832) f16
__global__ void k_build_xcat2(const float* __restrict__ x, const float* __restrict__ h,
                              const int* __restrict__ lab, const float* __restrict__ emb,
                              half_t* __restrict__ d)
{
    size_t i = (size_t)blockIdx.x * blockDim.x + threadIdx.x;
    if (i >= (size_t)NN_ * K2P_) return;
    int c = (int)(i % K2P_);
    int r = (int)(i / K2P_);
    float v = 0.f;
    if (c < CIN_)               v = x[(size_t)r * CIN_ + c];
    else if (c < CIN_ + DD_)    v = h[r * DD_ + (c - CIN_)];
    else if (c < K2_)           v = emb[lab[r] * EMB_ + (c - CIN_ - DD_)];
    d[i] = (half_t)v;
}

// dedup: pos2rep[p0*N+p1] = min edge index mapping there
__global__ void k_rep_min(const int* __restrict__ rp, int* __restrict__ rep)
{
    int e = blockIdx.x * blockDim.x + threadIdx.x;
    if (e >= EE_) return;
    atomicMin(&rep[rp[2 * e] * NN_ + rp[2 * e + 1]], e);
}

// deg[p0] += 1 for each unique edge
__global__ void k_deg(const int* __restrict__ rp, const int* __restrict__ rep,
                      float* __restrict__ deg)
{
    int e = blockIdx.x * blockDim.x + threadIdx.x;
    if (e >= EE_) return;
    int p0 = rp[2 * e], p1 = rp[2 * e + 1];
    if (rep[p0 * NN_ + p1] == e) atomicAdd(&deg[p0], 1.0f);
}

// esRep[rep(e)] += er[e]   (duplicate pairs accumulate, matching .at[].add)
__global__ void k_scatter_es(const int* __restrict__ rp, const int* __restrict__ rep,
                             const float* __restrict__ er, float* __restrict__ es)
{
    size_t i = (size_t)blockIdx.x * blockDim.x + threadIdx.x;
    if (i >= (size_t)EE_ * DD_) return;
    int e = (int)(i >> 9), d = (int)(i & 511);
    int r = rep[rp[2 * e] * NN_ + rp[2 * e + 1]];
    atomicAdd(&es[(size_t)r * DD_ + d], er[i]);
}

// hE[e] = esRep[rep(e)]  (broadcast positional state to duplicate edges)
__global__ void k_gather_h0(const int* __restrict__ rp, const int* __restrict__ rep,
                            const float* __restrict__ es, float* __restrict__ hE)
{
    size_t i = (size_t)blockIdx.x * blockDim.x + threadIdx.x;
    if (i >= (size_t)EE_ * DD_) return;
    int e = (int)(i >> 9), d = (int)(i & 511);
    int r = rep[rp[2 * e] * NN_ + rp[2 * e + 1]];
    hE[i] = es[(size_t)r * DD_ + d];
}

// e2n[p1] += hE[e] over unique edges
__global__ void k_e2n(const int* __restrict__ rp, const int* __restrict__ rep,
                      const float* __restrict__ hE, float* __restrict__ o)
{
    size_t i = (size_t)blockIdx.x * blockDim.x + threadIdx.x;
    if (i >= (size_t)EE_ * DD_) return;
    int e = (int)(i >> 9), d = (int)(i & 511);
    int p0 = rp[2 * e], p1 = rp[2 * e + 1];
    if (rep[p0 * NN_ + p1] == e) atomicAdd(&o[(size_t)p1 * DD_ + d], hE[i]);
}

// n2n[p0] += node_states[p1] over unique edges
__global__ void k_n2n_scatter(const int* __restrict__ rp, const int* __restrict__ rep,
                              const float* __restrict__ h, float* __restrict__ o)
{
    size_t i = (size_t)blockIdx.x * blockDim.x + threadIdx.x;
    if (i >= (size_t)EE_ * DD_) return;
    int e = (int)(i >> 9), d = (int)(i & 511);
    int p0 = rp[2 * e], p1 = rp[2 * e + 1];
    if (rep[p0 * NN_ + p1] == e) atomicAdd(&o[(size_t)p0 * DD_ + d], h[(size_t)p1 * DD_ + d]);
}

__global__ void k_n2n_div(float* __restrict__ o, const float* __restrict__ deg)
{
    int i = blockIdx.x * blockDim.x + threadIdx.x;
    if (i >= NN_ * DD_) return;
    o[i] /= (deg[i >> 9] + 1e-6f);
}

// inp = 0.5*(a+b)   (ALPHA = 0.5)
__global__ void k_combine(const float* __restrict__ a, const float* __restrict__ b,
                          float* __restrict__ o, int n)
{
    int i = blockIdx.x * blockDim.x + threadIdx.x;
    if (i < n) o[i] = 0.5f * (a[i] + b[i]);
}

// GRU gate: h = (1-z)*tanh(ig + r*hg) + z*h   (gi/gh rows of 1536, h rows of 512)
__global__ void k_gru_gate(const float* __restrict__ gi, const float* __restrict__ gh,
                           float* __restrict__ h, int rows)
{
    size_t i = (size_t)blockIdx.x * blockDim.x + threadIdx.x;
    if (i >= (size_t)rows * DD_) return;
    int r = (int)(i >> 9), d = (int)(i & 511);
    const float* gir = gi + (size_t)r * 1536;
    const float* ghr = gh + (size_t)r * 1536;
    float rr = 1.0f / (1.0f + expf(-(gir[d]        + ghr[d])));
    float zz = 1.0f / (1.0f + expf(-(gir[512 + d]  + ghr[512 + d])));
    float gg = tanhf(gir[1024 + d] + rr * ghr[1024 + d]);
    h[i] = (1.0f - zz) * gg + zz * h[i];
}

// ek[e] = [ns2[p0] | ns2[p1]]  (E x 1024) f16  (adj==1 at all edges)
__global__ void k_build_ek(const int* __restrict__ rp, const float* __restrict__ ns,
                           half_t* __restrict__ d)
{
    size_t i = (size_t)blockIdx.x * blockDim.x + threadIdx.x;
    if (i >= (size_t)EE_ * 1024) return;
    int e = (int)(i >> 10), c = (int)(i & 1023);
    float v = (c < DD_) ? ns[(size_t)rp[2 * e] * DD_ + c]
                        : ns[(size_t)rp[2 * e + 1] * DD_ + (c - DD_)];
    d[i] = (half_t)v;
}

// obj_dists: -1000 everywhere, +1000 at label
__global__ void k_obj_dists(const int* __restrict__ lab, float* __restrict__ o)
{
    int i = blockIdx.x * blockDim.x + threadIdx.x;
    if (i >= NN_ * NOBJ_) return;
    int r = i / NOBJ_, c = i % NOBJ_;
    o[i] = (c == lab[r]) ? 1000.0f : -1000.0f;
}

// ---------------------------------------------------------------------------
extern "C" void kernel_launch(void* const* d_in, const int* in_sizes, int n_in,
                              void* d_out, int out_size, void* d_ws, size_t ws_size,
                              hipStream_t stream)
{
    (void)in_sizes; (void)n_in; (void)out_size; (void)ws_size;

    const float* x      = (const float*)d_in[0];
    const float* pinfo  = (const float*)d_in[1];
    const float* uni    = (const float*)d_in[2];
    const int*   lab    = (const int*)d_in[3];
    const int*   rp     = (const int*)d_in[4];
    const float* emb1   = (const float*)d_in[5];
    const float* emb2   = (const float*)d_in[6];
    const float* w_n1   = (const float*)d_in[7];
    const float* b_n1   = (const float*)d_in[8];
    const float* w_n2   = (const float*)d_in[9];
    const float* b_n2   = (const float*)d_in[10];
    const float* w_ed   = (const float*)d_in[11];
    const float* b_ed   = (const float*)d_in[12];
    const float* pw1    = (const float*)d_in[13];
    const float* pb1    = (const float*)d_in[14];
    const float* pg     = (const float*)d_in[15];
    const float* pbta   = (const float*)d_in[16];
    const float* pw2    = (const float*)d_in[17];
    const float* pb2    = (const float*)d_in[18];
    const float* ngih   = (const float*)d_in[19];
    const float* nghh   = (const float*)d_in[20];
    const float* ngbih  = (const float*)d_in[21];
    const float* ngbhh  = (const float*)d_in[22];
    const float* egih   = (const float*)d_in[23];
    const float* eghh   = (const float*)d_in[24];
    const float* n2n1w  = (const float*)d_in[25];
    const float* n2n1b  = (const float*)d_in[26];
    const float* n2n2w  = (const float*)d_in[27];
    const float* n2n2b  = (const float*)d_in[28];
    const float* e2n1w  = (const float*)d_in[29];
    const float* e2n1b  = (const float*)d_in[30];
    const float* e2n2w  = (const float*)d_in[31];
    const float* e2n2b  = (const float*)d_in[32];
    const float* n2e1w  = (const float*)d_in[33];
    const float* n2e1b  = (const float*)d_in[34];
    const float* n2e2w  = (const float*)d_in[35];
    const float* n2e2b  = (const float*)d_in[36];
    const float* relw   = (const float*)d_in[37];
    const float* relb   = (const float*)d_in[38];
    float* out = (float*)d_out;

    // ---- bump allocator over d_ws ----
    char* ws = (char*)d_ws; size_t off = 0;
    auto alloc = [&](size_t bytes) -> void* {
        void* p = ws + off; off = (off + bytes + 255) & ~(size_t)255; return p;
    };
    // f16 weights (B operands, rows padded to 128 where needed)
    half_t* w_n1h  = (half_t*)alloc((size_t)DD_ * K1P_ * 2);
    half_t* w_edh  = (half_t*)alloc((size_t)DD_ * CIN_ * 2);
    half_t* w_n2h  = (half_t*)alloc((size_t)DD_ * K2P_ * 2);
    half_t* w_ngih = (half_t*)alloc((size_t)1536 * DD_ * 2);
    half_t* w_nghh = (half_t*)alloc((size_t)1536 * DD_ * 2);
    half_t* w_egih = (half_t*)alloc((size_t)1536 * DD_ * 2);
    half_t* w_eghh = (half_t*)alloc((size_t)1536 * DD_ * 2);
    half_t* w_a1   = (half_t*)alloc((size_t)DD_ * DD_ * 2);   // n2n_w1
    half_t* w_a2   = (half_t*)alloc((size_t)DD_ * DD_ * 2);   // n2n_w2
    half_t* w_b1   = (half_t*)alloc((size_t)DD_ * DD_ * 2);   // e2n_w1
    half_t* w_b2   = (half_t*)alloc((size_t)DD_ * DD_ * 2);   // e2n_w2
    half_t* w_c1   = (half_t*)alloc((size_t)DD_ * 1024 * 2);  // n2e_w1
    half_t* w_c2   = (half_t*)alloc((size_t)DD_ * DD_ * 2);   // n2e_w2
    half_t* w_relh = (half_t*)alloc((size_t)128 * DD_ * 2);   // 51 -> 128 rows
    // activations / scratch
    float*  ph     = (float*)alloc((size_t)NN_ * 32 * 4);
    float*  mu     = (float*)alloc(32 * 4);
    float*  rsig   = (float*)alloc(32 * 4);
    float*  phn    = (float*)alloc((size_t)NN_ * 32 * 4);
    float*  pose   = (float*)alloc((size_t)NN_ * 128 * 4);
    half_t* xcat   = (half_t*)alloc((size_t)NN_ * K1P_ * 2);
    float*  hN     = (float*)alloc((size_t)NN_ * DD_ * 4);    // node_states
    half_t* hNh    = (half_t*)alloc((size_t)NN_ * DD_ * 2);
    half_t* unih   = (half_t*)alloc((size_t)EE_ * CIN_ * 2);
    float*  er     = (float*)alloc((size_t)EE_ * DD_ * 4);
    int*    rep    = (int*)alloc((size_t)NN_ * NN_ * 4);
    float*  esRep  = (float*)alloc((size_t)EE_ * DD_ * 4);
    float*  hE     = (float*)alloc((size_t)EE_ * DD_ * 4);
    half_t* hEh    = (half_t*)alloc((size_t)EE_ * DD_ * 2);
    float*  degf   = (float*)alloc(NN_ * 4);
    float*  e2nr   = (float*)alloc((size_t)NN_ * DD_ * 4);
    float*  n2nr   = (float*)alloc((size_t)NN_ * DD_ * 4);
    half_t* m512h  = (half_t*)alloc((size_t)NN_ * DD_ * 2);
    float*  tmpA   = (float*)alloc((size_t)NN_ * DD_ * 4);
    half_t* tmpAh  = (half_t*)alloc((size_t)NN_ * DD_ * 2);
    float*  n2nm   = (float*)alloc((size_t)NN_ * DD_ * 4);
    float*  e2nm   = (float*)alloc((size_t)NN_ * DD_ * 4);
    float*  inp    = (float*)alloc((size_t)NN_ * DD_ * 4);
    half_t* inph   = (half_t*)alloc((size_t)NN_ * DD_ * 2);
    float*  giN    = (float*)alloc((size_t)NN_ * 1536 * 4);
    float*  ghN    = (float*)alloc((size_t)NN_ * 1536 * 4);
    half_t* xcat2  = (half_t*)alloc((size_t)NN_ * K2P_ * 2);
    float*  ns2    = (float*)alloc((size_t)NN_ * DD_ * 4);
    half_t* ekh    = (half_t*)alloc((size_t)EE_ * 1024 * 2);
    float*  t1     = (float*)alloc((size_t)EE_ * DD_ * 4);
    half_t* t1h    = (half_t*)alloc((size_t)EE_ * DD_ * 2);
    float*  msg    = (float*)alloc((size_t)EE_ * DD_ * 4);
    half_t* msgh   = (half_t*)alloc((size_t)EE_ * DD_ * 2);
    float*  giE    = (float*)alloc((size_t)EE_ * 1536 * 4);
    float*  ghE    = (float*)alloc((size_t)EE_ * 1536 * 4);

    auto conv = [&](const float* s, half_t* d, int Mr, int Mp, int K, int Kp) {
        size_t tot = (size_t)Mp * Kp;
        k_f32_to_f16<<<dim3((unsigned)((tot + 255) / 256)), dim3(256), 0, stream>>>(s, d, Mr, Mp, K, Kp);
    };
    auto zero = [&](float* p, size_t n) {
        k_zero_f32<<<dim3((unsigned)((n + 255) / 256)), dim3(256), 0, stream>>>(p, n);
    };
    // Npad: B row count (mult of 128, zero padded); Nout: real/stored columns
    auto gemm = [&](const half_t* A, const half_t* B, const float* bias, float* C,
                    int M, int Nout, int Npad, int K, int ldc, int act) {
        dim3 g(Npad / TN, M / TM);
        gemm_f16<<<g, dim3(256), 0, stream>>>(A, B, bias, C, M, Nout, K, ldc, act);
    };

    // ---- weight conversion (f32 -> f16, K- and row-padded) ----
    conv(w_n1, w_n1h, DD_, DD_, K1_, K1P_);
    conv(w_ed, w_edh, DD_, DD_, CIN_, CIN_);
    conv(w_n2, w_n2h, DD_, DD_, K2_, K2P_);
    conv(ngih, w_ngih, 1536, 1536, DD_, DD_);
    conv(nghh, w_nghh, 1536, 1536, DD_, DD_);
    conv(egih, w_egih, 1536, 1536, DD_, DD_);
    conv(eghh, w_eghh, 1536, 1536, DD_, DD_);
    conv(n2n1w, w_a1, DD_, DD_, DD_, DD_);
    conv(n2n2w, w_a2, DD_, DD_, DD_, DD_);
    conv(e2n1w, w_b1, DD_, DD_, DD_, DD_);
    conv(e2n2w, w_b2, DD_, DD_, DD_, DD_);
    conv(n2e1w, w_c1, DD_, DD_, 1024, 1024);
    conv(n2e2w, w_c2, DD_, DD_, DD_, DD_);
    conv(relw, w_relh, NREL_, 128, DD_, DD_);

    // ---- positional embedding (tiny, scalar) ----
    k_pos1<<<dim3(cdiv_h(NN_ * 32, 256)), dim3(256), 0, stream>>>(pinfo, pw1, pb1, ph);
    k_colstats<<<dim3(1), dim3(32), 0, stream>>>(ph, mu, rsig);
    k_posnorm<<<dim3(cdiv_h(NN_ * 32, 256)), dim3(256), 0, stream>>>(ph, mu, rsig, pg, pbta, phn);
    k_pos2<<<dim3(cdiv_h(NN_ * 128, 256)), dim3(256), 0, stream>>>(phn, pw2, pb2, pose);

    // ---- node_states = Xcat @ W1^T + b ----
    {
        size_t tot = (size_t)NN_ * K1P_;
        k_build_xcat<<<dim3((unsigned)((tot + 255) / 256)), dim3(256), 0, stream>>>(x, lab, emb1, pose, xcat);
    }
    gemm(xcat, w_n1h, b_n1, hN, NN_, DD_, DD_, K1P_, DD_, 0);

    // ---- edge_rep = union @ We^T + b  (13 GFLOP GEMM) ----
    conv(uni, unih, EE_, EE_, CIN_, CIN_);
    gemm(unih, w_edh, b_ed, er, EE_, DD_, DD_, CIN_, DD_, 0);

    // ---- dedup pairs, degree, sparse edge state ----
    {
        size_t nn = (size_t)NN_ * NN_;
        k_fill_i32<<<dim3((unsigned)((nn + 255) / 256)), dim3(256), 0, stream>>>(rep, 0x7FFFFFFF, nn);
    }
    k_rep_min<<<dim3(cdiv_h(EE_, 256)), dim3(256), 0, stream>>>(rp, rep);
    zero(degf, NN_);
    k_deg<<<dim3(cdiv_h(EE_, 256)), dim3(256), 0, stream>>>(rp, rep, degf);
    zero(esRep, (size_t)EE_ * DD_);
    {
        unsigned g = (unsigned)(((size_t)EE_ * DD_ + 255) / 256);
        k_scatter_es<<<dim3(g), dim3(256), 0, stream>>>(rp, rep, er, esRep);
        k_gather_h0<<<dim3(g), dim3(256), 0, stream>>>(rp, rep, esRep, hE);
        // e2n is loop-invariant in phase 1 (edge_states constant): compute once
        zero(e2nr, (size_t)NN_ * DD_);
        k_e2n<<<dim3(g), dim3(256), 0, stream>>>(rp, rep, hE, e2nr);
    }
    conv(e2nr, m512h, NN_, NN_, DD_, DD_);
    gemm(m512h, w_b1, e2n1b, tmpA, NN_, DD_, DD_, DD_, DD_, 1);
    conv(tmpA, tmpAh, NN_, NN_, DD_, DD_);
    gemm(tmpAh, w_b2, e2n2b, e2nm, NN_, DD_, DD_, DD_, DD_, 1);

    // ---- phase-1 node GRU loop ----
    for (int it = 0; it < 3; ++it) {
        zero(n2nr, (size_t)NN_ * DD_);
        {
            unsigned g = (unsigned)(((size_t)EE_ * DD_ + 255) / 256);
            k_n2n_scatter<<<dim3(g), dim3(256), 0, stream>>>(rp, rep, hN, n2nr);
        }
        k_n2n_div<<<dim3(cdiv_h(NN_ * DD_, 256)), dim3(256), 0, stream>>>(n2nr, degf);
        conv(n2nr, m512h, NN_, NN_, DD_, DD_);
        gemm(m512h, w_a1, n2n1b, tmpA, NN_, DD_, DD_, DD_, DD_, 1);
        conv(tmpA, tmpAh, NN_, NN_, DD_, DD_);
        gemm(tmpAh, w_a2, n2n2b, n2nm, NN_, DD_, DD_, DD_, DD_, 1);
        k_combine<<<dim3(cdiv_h(NN_ * DD_, 256)), dim3(256), 0, stream>>>(n2nm, e2nm, inp, NN_ * DD_);
        conv(inp, inph, NN_, NN_, DD_, DD_);
        gemm(inph, w_ngih, ngbih, giN, NN_, 1536, 1536, DD_, 1536, 0);
        conv(hN, hNh, NN_, NN_, DD_, DD_);
        gemm(hNh, w_nghh, ngbhh, ghN, NN_, 1536, 1536, DD_, 1536, 0);
        k_gru_gate<<<dim3(cdiv_h(NN_ * DD_, 256)), dim3(256), 0, stream>>>(giN, ghN, hN, NN_);
    }

    // ---- node_states2 = Xcat2 @ W2^T + b ----
    {
        size_t tot = (size_t)NN_ * K2P_;
        k_build_xcat2<<<dim3((unsigned)((tot + 255) / 256)), dim3(256), 0, stream>>>(x, hN, lab, emb2, xcat2);
    }
    gemm(xcat2, w_n2h, b_n2, ns2, NN_, DD_, DD_, K2P_, DD_, 0);

    // ---- phase-2 edge GRU: ek/msg/gi are loop-invariant ----
    {
        size_t tot = (size_t)EE_ * 1024;
        k_build_ek<<<dim3((unsigned)((tot + 255) / 256)), dim3(256), 0, stream>>>(rp, ns2, ekh);
    }
    gemm(ekh, w_c1, n2e1b, t1, EE_, DD_, DD_, 1024, DD_, 1);
    conv(t1, t1h, EE_, EE_, DD_, DD_);
    gemm(t1h, w_c2, n2e2b, msg, EE_, DD_, DD_, DD_, DD_, 1);
    conv(msg, msgh, EE_, EE_, DD_, DD_);
    gemm(msgh, w_egih, nullptr, giE, EE_, 1536, 1536, DD_, 1536, 0);   // egate: no bias

    for (int it = 0; it < 3; ++it) {
        conv(hE, hEh, EE_, EE_, DD_, DD_);
        gemm(hEh, w_eghh, nullptr, ghE, EE_, 1536, 1536, DD_, 1536, 0);
        {
            unsigned g = (unsigned)(((size_t)EE_ * DD_ + 255) / 256);
            k_gru_gate<<<dim3(g), dim3(256), 0, stream>>>(giE, ghE, hE, EE_);
        }
    }

    // ---- outputs ----
    k_obj_dists<<<dim3(cdiv_h(NN_ * NOBJ_, 256)), dim3(256), 0, stream>>>(lab, out);
    conv(hE, hEh, EE_, EE_, DD_, DD_);
    gemm(hEh, w_relh, relb, out + (size_t)NN_ * NOBJ_, EE_, NREL_, 128, DD_, NREL_, 0);
}
